// GAT_45397804319027
// MI455X (gfx1250) — compile-verified, matrix operand order checked by reading, not run on hardware
//
#include <hip/hip_runtime.h>
#include <math.h>

// ---------------------------------------------------------------------------
// GAT forward, 2 layers, gfx1250 (CDNA5).  bf16 WMMA for the dense GEMMs
// (B pre-packed into WMMA fragment layout), f32 atomics for the sparse
// segment-softmax / aggregation.
// ---------------------------------------------------------------------------

typedef __attribute__((ext_vector_type(16))) __bf16 v16bf;
typedef __attribute__((ext_vector_type(8)))  float  v8f;

#define LEAKY_SLOPE 0.2f

// ---------------- generic fill ----------------
__global__ void fill_kernel(float* __restrict__ p, float v, int n) {
    int i = blockIdx.x * blockDim.x + threadIdx.x;
    if (i < n) p[i] = v;
}

// ---------------- pack W1 [H,F,D] -> [F, H*D] ----------------
__global__ void pack_w1_kernel(const float* __restrict__ W, float* __restrict__ Wc,
                               int F, int H, int D) {
    int i = blockIdx.x * blockDim.x + threadIdx.x;
    int tot = F * H * D;
    if (i >= tot) return;
    int f = i / (H * D);
    int r = i - f * (H * D);
    int h = r / D;
    int d = r - h * D;
    Wc[f * (H * D) + h * D + d] = W[(h * F + f) * D + d];
}

// ---------------- pack B [K,N] f32 -> WMMA bf16 fragment layout -------------
// Element i = ((ks*NT + t)*32 + lane)*16 + e, where per 16-bit B 32x16
// fragment: lane -> col = t*16 + (lane&15), half = lane>>4; e -> vgpr j=e>>1,
// k = ks*32 + ((j<4)?0:16) + 8*half + 2*(j&3) + (e&1).  Zero-padded for n>=N.
__global__ void pack_b_wmma_kernel(const float* __restrict__ B, __bf16* __restrict__ Bp,
                                   int K, int N, int NT) {
    int i = blockIdx.x * blockDim.x + threadIdx.x;
    int ksteps = K >> 5;
    int tot = NT * ksteps * 32 * 16;
    if (i >= tot) return;
    int e    = i & 15;
    int lane = (i >> 4) & 31;
    int rest = i >> 9;            // ks*NT + t
    int t    = rest % NT;
    int ks   = rest / NT;
    int half = lane >> 4;
    int n    = t * 16 + (lane & 15);
    int j    = e >> 1;
    int k    = ks * 32 + ((j < 4) ? 0 : 16) + 8 * half + 2 * (j & 3) + (e & 1);
    float v = (n < N) ? B[(size_t)k * N + n] : 0.f;
    Bp[i] = (__bf16)v;
}

// ---------------- WMMA GEMM: C[M,N] = A[M,K] * B(packed) --------------------
// One wave owns a 16-row stripe and NT column tiles; A streamed once, f32 ->
// bf16 converted in registers, B read pre-packed.  M multiple of 16.
template <int NT, int K>
__global__ void gemm_bf16_wmma(const float* __restrict__ A, const __bf16* __restrict__ Bp,
                               float* __restrict__ C, int M, int N) {
    const int wave  = threadIdx.x >> 5;
    const int lane  = threadIdx.x & 31;
    const int mtile = blockIdx.x * (blockDim.x >> 5) + wave;
    const int row0  = mtile * 16;
    if (row0 >= M) return;                 // uniform per wave -> EXEC all-1 inside

    const int half = lane >> 4;
    const int ncol = lane & 15;
    constexpr int KSTEPS = K / 32;

    v8f acc[NT];
#pragma unroll
    for (int t = 0; t < NT; ++t) acc[t] = (v8f){0.f,0.f,0.f,0.f,0.f,0.f,0.f,0.f};

    // A fragment rows: lane&15; K-pair runs start at 8*half and 16+8*half
    const float*  __restrict__ arow = A + (size_t)(row0 + (lane & 15)) * K + 8 * half;
    const __bf16* __restrict__ bp   = Bp + (size_t)lane * 16;

#pragma unroll 4
    for (int ks = 0; ks < KSTEPS; ++ks) {
        const float4 r0 = *(const float4*)(arow + ks * 32);
        const float4 r1 = *(const float4*)(arow + ks * 32 + 4);
        const float4 r2 = *(const float4*)(arow + ks * 32 + 16);
        const float4 r3 = *(const float4*)(arow + ks * 32 + 20);
        v16bf afrag;
        afrag[0]  = (__bf16)r0.x; afrag[1]  = (__bf16)r0.y;
        afrag[2]  = (__bf16)r0.z; afrag[3]  = (__bf16)r0.w;
        afrag[4]  = (__bf16)r1.x; afrag[5]  = (__bf16)r1.y;
        afrag[6]  = (__bf16)r1.z; afrag[7]  = (__bf16)r1.w;
        afrag[8]  = (__bf16)r2.x; afrag[9]  = (__bf16)r2.y;
        afrag[10] = (__bf16)r2.z; afrag[11] = (__bf16)r2.w;
        afrag[12] = (__bf16)r3.x; afrag[13] = (__bf16)r3.y;
        afrag[14] = (__bf16)r3.z; afrag[15] = (__bf16)r3.w;
#pragma unroll
        for (int t = 0; t < NT; ++t) {
            const v16bf bfrag = *(const v16bf*)(bp + (size_t)(ks * NT + t) * 512);
            acc[t] = __builtin_amdgcn_wmma_f32_16x16x32_bf16(
                false, afrag, false, bfrag, (short)0, acc[t], false, false);
        }
    }

    // f32 C/D layout: VGPR r -> row = r + 8*half, col = lane&15
#pragma unroll
    for (int t = 0; t < NT; ++t) {
        const int col = t * 16 + ncol;
        if (col < N) {
#pragma unroll
            for (int r = 0; r < 8; ++r) {
                C[(size_t)(row0 + r + 8 * half) * N + col] = acc[t][r];
            }
        }
    }
}

// ---------------- per-(node,head) attention scores ----------------
__global__ void scores_kernel(const float* __restrict__ h, const float* __restrict__ a,
                              float* __restrict__ s_src, float* __restrict__ s_dst,
                              int N, int H, int D) {
    int i = blockIdx.x * blockDim.x + threadIdx.x;
    if (i >= N * H) return;
    int n  = i / H;
    int hd = i - n * H;
    const float* hp = h + ((size_t)n * H + hd) * D;
    const float* ap = a + (size_t)hd * 2 * D;
    float ss = 0.f, sd = 0.f;
    for (int d = 0; d < D; ++d) {
        ss += hp[d] * ap[d];
        sd += hp[d] * ap[D + d];
    }
    s_src[i] = ss;
    s_dst[i] = sd;
}

// ---------------- float atomic max via int/uint ordering trick ----------------
__device__ __forceinline__ void atomicMaxF(float* addr, float val) {
    if (val >= 0.f) atomicMax((int*)addr, __float_as_int(val));
    else            atomicMin((unsigned int*)addr, __float_as_uint(val));
}

// pass 1: e = leaky_relu(s_src[src] + s_dst[dst]); m[src] = max(m[src], e)
__global__ void edge_max_kernel(const int* __restrict__ src, const int* __restrict__ dst,
                                const float* __restrict__ ss, const float* __restrict__ sd,
                                float* __restrict__ m, int E, int H) {
    int i = blockIdx.x * blockDim.x + threadIdx.x;
    if (i >= E * H) return;
    int e  = i / H;
    int hd = i - e * H;
    int s = src[e], d = dst[e];
    float v = ss[s * H + hd] + sd[d * H + hd];
    v = (v > 0.f) ? v : LEAKY_SLOPE * v;
    atomicMaxF(&m[s * H + hd], v);
}

// pass 2: ex = exp(e - m[src]); denom[src] += ex
__global__ void edge_exp_kernel(const int* __restrict__ src, const int* __restrict__ dst,
                                const float* __restrict__ ss, const float* __restrict__ sd,
                                const float* __restrict__ m, float* __restrict__ ex,
                                float* __restrict__ denom, int E, int H) {
    int i = blockIdx.x * blockDim.x + threadIdx.x;
    if (i >= E * H) return;
    int e  = i / H;
    int hd = i - e * H;
    int s = src[e], d = dst[e];
    float v = ss[s * H + hd] + sd[d * H + hd];
    v = (v > 0.f) ? v : LEAKY_SLOPE * v;
    float x = __expf(v - m[s * H + hd]);
    ex[i] = x;
    atomicAdd(&denom[s * H + hd], x);
}

// pass 3: out[src] += (ex/denom[src]) * h[dst]
__global__ void edge_agg_kernel(const int* __restrict__ src, const int* __restrict__ dst,
                                const float* __restrict__ ex, const float* __restrict__ denom,
                                const float* __restrict__ h, float* __restrict__ out,
                                int E, int H, int D) {
    int i = blockIdx.x * blockDim.x + threadIdx.x;
    if (i >= E * H) return;
    int e  = i / H;
    int hd = i - e * H;
    int s = src[e], d = dst[e];
    float alpha = ex[i] / denom[s * H + hd];
    const float* hp = h   + ((size_t)d * H + hd) * D;
    float*       op = out + ((size_t)s * H + hd) * D;
    for (int k = 0; k < D; ++k) atomicAdd(&op[k], alpha * hp[k]);
}

// ---------------- ELU (in place) ----------------
__global__ void elu_kernel(float* __restrict__ p, int n) {
    int i = blockIdx.x * blockDim.x + threadIdx.x;
    if (i >= n) return;
    float x = p[i];
    p[i] = (x > 0.f) ? x : (__expf(x) - 1.f);
}

// ---------------------------------------------------------------------------
extern "C" void kernel_launch(void* const* d_in, const int* in_sizes, int n_in,
                              void* d_out, int out_size, void* d_ws, size_t ws_size,
                              hipStream_t stream) {
    const float* x   = (const float*)d_in[0];          // [N, 512]
    const int*   ei  = (const int*)d_in[1];            // [2, E]
    const float* W1  = (const float*)d_in[2];          // [8, 512, 8]
    const float* a1  = (const float*)d_in[3];          // [8, 16]
    const float* W2  = (const float*)d_in[4];          // [1, 64, 40] == [64, 40]
    const float* a2  = (const float*)d_in[5];          // [1, 80]

    const int NFEAT = 512, NHID = 64, NCLASS = 40, H1 = 8, D1 = 8;
    const int N = in_sizes[0] / NFEAT;                 // 100000
    const int E = in_sizes[1] / 2;                     // 1600000
    const int* src = ei;
    const int* dst = ei + E;

    // ---- workspace carve-up (f32 units; bf16 packs carved in float units) ----
    float* ws = (float*)d_ws;
    size_t off = 0;
    float* h1   = ws + off; off += (size_t)N * NHID;       // [N,64]
    float* Wc1  = ws + off; off += (size_t)NFEAT * NHID;   // [512,64]
    float* ss1  = ws + off; off += (size_t)N * H1;
    float* sd1  = ws + off; off += (size_t)N * H1;
    float* m1   = ws + off; off += (size_t)N * H1;
    float* dn1  = ws + off; off += (size_t)N * H1;
    float* ex1  = ws + off; off += (size_t)E * H1;         // [E,8]
    float* agg1 = ws + off; off += (size_t)N * NHID;       // [N,64]
    float* h2   = ws + off; off += (size_t)N * NCLASS;     // [N,40]
    float* ss2  = ws + off; off += (size_t)N;
    float* sd2  = ws + off; off += (size_t)N;
    float* m2   = ws + off; off += (size_t)N;
    float* dn2  = ws + off; off += (size_t)N;
    float* ex2  = ws + off; off += (size_t)E;
    // packed B fragments (bf16): layer1 NT=4,ks=16 -> 32768; layer2 NT=3,ks=2 -> 3072
    __bf16* bp1 = (__bf16*)(ws + off); off += 16384;       // 32768 bf16
    __bf16* bp2 = (__bf16*)(ws + off); off += 2048;        // 3072 bf16 (padded)

    const int TB = 256;
    auto blocks = [](long long n, int tb) { return (int)((n + tb - 1) / tb); };

    // ================= layer 1 =================
    fill_kernel<<<blocks((long long)N * H1, TB), TB, 0, stream>>>(m1, -INFINITY, N * H1);
    fill_kernel<<<blocks((long long)N * H1, TB), TB, 0, stream>>>(dn1, 0.f, N * H1);
    fill_kernel<<<blocks((long long)N * NHID, TB), TB, 0, stream>>>(agg1, 0.f, N * NHID);
    pack_w1_kernel<<<blocks((long long)NFEAT * NHID, TB), TB, 0, stream>>>(W1, Wc1, NFEAT, H1, D1);
    pack_b_wmma_kernel<<<blocks(4LL * 16 * 32 * 16, TB), TB, 0, stream>>>(Wc1, bp1, NFEAT, NHID, 4);

    {   // h1 = x @ Wc1 : [100000,512]x[512,64], NT=4, K=512
        int mtiles = N / 16;
        int wavesPerBlock = TB / 32;
        gemm_bf16_wmma<4, 512><<<blocks(mtiles, wavesPerBlock), TB, 0, stream>>>(
            x, bp1, h1, N, NHID);
    }
    scores_kernel<<<blocks((long long)N * H1, TB), TB, 0, stream>>>(h1, a1, ss1, sd1, N, H1, D1);
    edge_max_kernel<<<blocks((long long)E * H1, TB), TB, 0, stream>>>(src, dst, ss1, sd1, m1, E, H1);
    edge_exp_kernel<<<blocks((long long)E * H1, TB), TB, 0, stream>>>(src, dst, ss1, sd1, m1, ex1, dn1, E, H1);
    edge_agg_kernel<<<blocks((long long)E * H1, TB), TB, 0, stream>>>(src, dst, ex1, dn1, h1, agg1, E, H1, D1);
    elu_kernel<<<blocks((long long)N * NHID, TB), TB, 0, stream>>>(agg1, N * NHID);

    // ================= layer 2 =================
    fill_kernel<<<blocks(N, TB), TB, 0, stream>>>(m2, -INFINITY, N);
    fill_kernel<<<blocks(N, TB), TB, 0, stream>>>(dn2, 0.f, N);
    fill_kernel<<<blocks((long long)N * NCLASS, TB), TB, 0, stream>>>((float*)d_out, 0.f, N * NCLASS);
    pack_b_wmma_kernel<<<blocks(3LL * 2 * 32 * 16, TB), TB, 0, stream>>>(W2, bp2, NHID, NCLASS, 3);

    {   // h2 = agg1 @ W2 : [100000,64]x[64,40], NT=3, K=64
        int mtiles = N / 16;
        int wavesPerBlock = TB / 32;
        gemm_bf16_wmma<3, 64><<<blocks(mtiles, wavesPerBlock), TB, 0, stream>>>(
            agg1, bp2, h2, N, NCLASS);
    }
    scores_kernel<<<blocks(N, TB), TB, 0, stream>>>(h2, a2, ss2, sd2, N, 1, NCLASS);
    edge_max_kernel<<<blocks(E, TB), TB, 0, stream>>>(src, dst, ss2, sd2, m2, E, 1);
    edge_exp_kernel<<<blocks(E, TB), TB, 0, stream>>>(src, dst, ss2, sd2, m2, ex2, dn2, E, 1);
    edge_agg_kernel<<<blocks(E, TB), TB, 0, stream>>>(src, dst, ex2, dn2, h2, (float*)d_out, E, 1, NCLASS);
}
